// LQTEncoder_4750233830102
// MI455X (gfx1250) — matrix operand order repaired and checked by reading, not compiled, not run on hardware
//
#include <hip/hip_runtime.h>
#include <math.h>

// ---------------------------------------------------------------------------
// LQT encoder for MI455X (gfx1250, wave32).
// Big GEMMs: v_wmma_f32_16x16x32_bf16, 128x64 tile, BK=32, 8 waves,
//            double-buffered LDS; A tile staged via
//            global_load_async_to_lds_b128 overlapped with WMMA compute;
//            fragments gathered as ds_load_b128 from packed LDS layouts.
// Attention (3% of FLOPs) + LayerNorm are VALU/LDS kernels.
// ---------------------------------------------------------------------------

typedef __bf16 bf16_t;
typedef __bf16 v16bf  __attribute__((ext_vector_type(16)));
typedef float  v8f    __attribute__((ext_vector_type(8)));
typedef __bf16 bf16x2 __attribute__((ext_vector_type(2)));
typedef __bf16 bf16x8 __attribute__((ext_vector_type(8)));
typedef unsigned int u32x4 __attribute__((ext_vector_type(4)));

#define BQ    64     // NQ
#define DIM   512
#define LSEQ  77
#define TXTD  768
#define NB    1024   // B
#define NH    8
#define HEADD 64
#define MLPD  1024

__device__ __forceinline__ float wave_sum(float v) {
  #pragma unroll
  for (int off = 16; off > 0; off >>= 1) v += __shfl_xor(v, off, 32);
  return v;
}

union V16 { u32x4 q[2]; v16bf v; };   // trivially-constructible bitcast helper

// ---------------------------------------------------------------------------
// WMMA GEMM: C[M,N] = epilogue(A[M,K] @ W[K,N] + bias) (+ residual)
// Tile 128x64, BK=32, double-buffered, last iteration peeled so the inner
// loop is branch-free. Wave w owns M-strip w*16 and all 64 N cols (4 accums).
// LDS layouts (packed so every fragment = two ds_load_b128):
//   sA: row-major bf16, pitch 20 dwords (32 elems + 4 dword pad)
//   sB: per-column K-pair packed: dword (col*20 + k/2) holds {k, k+1} bf16.
// bf16 A is staged by the async DMA path (per-lane LDS addresses keep pad)
// and overlapped with compute on the other buffer.
// Requires N%64==0, K%32==0; M%128==0 on the bf16-A path (true here),
// f32-A path is row-guarded (mq GEMM has M=64).
// ---------------------------------------------------------------------------
template<typename AT, typename OutT, bool GELU>
__global__ __launch_bounds__(256)
void gemm_wmma(const AT* __restrict__ A, const float* __restrict__ W,
               const float* __restrict__ bias, const float* __restrict__ residual,
               OutT* __restrict__ C, int M, int N, int K)
{
  constexpr int BM = 128, BN = 64, BK = 32;
  constexpr int LDAU = 20;                 // dwords per A row    (16 + 4 pad)
  constexpr int LDBU = 20;                 // dwords per B column (16 + 4 pad)
  constexpr bool kAsyncA = __is_same(AT, bf16_t);

  __shared__ unsigned int sAu[2][BM * LDAU];  // 2 x 10240 B
  __shared__ unsigned int sBu[2][BN * LDBU];  // 2 x  5120 B

  const int tid  = threadIdx.x;
  const int bm   = blockIdx.x * BM;
  const int bn   = blockIdx.y * BN;
  const int mi   = tid >> 5;               // wave id -> 16-row M strip
  const int lane = tid & 31;
  const int lh   = lane >> 4;              // ISA fragment lane-half
  const int l16  = lane & 15;

  // staging coordinates
  const int arow = tid >> 1;               // 0..127
  const int acb  = (tid & 1) * 16;         // 16 bf16 (32 B) per thread
  const int brow = tid >> 3;               // 0..31 (K row of W tile)
  const int bcb  = (tid & 7) * 8;          // 8 f32 per thread

  const unsigned int ldsA0 = (unsigned int)(unsigned long long)
      (&((bf16_t*)sAu[0])[arow * (LDAU * 2) + acb]);

  // f32-A path: rows past M are zeroed once and never overwritten.
  if constexpr (!kAsyncA) {
    if (bm + arow >= M) {
      #pragma unroll
      for (int buf = 0; buf < 2; ++buf) {
        bf16_t* dst = (bf16_t*)sAu[buf] + arow * (LDAU * 2) + acb;
        #pragma unroll
        for (int j = 0; j < 16; ++j) dst[j] = (bf16_t)0.0f;
      }
    }
  }

  // stage one 128x32 A tile + 32x64 W tile into LDS buffer `buf`
  auto stage = [&](int buf, int k0) {
    if constexpr (kAsyncA) {
      const bf16_t* src = A + (size_t)(bm + arow) * K + k0 + acb;
      const unsigned int la = ldsA0 + (unsigned int)buf * (BM * LDAU * 4);
      // DMA 2x16B straight into padded LDS; INST_OFFSET shifts both addresses.
      asm volatile("global_load_async_to_lds_b128 %0, %1, off"
                   :: "v"(la), "v"(src) : "memory");
      asm volatile("global_load_async_to_lds_b128 %0, %1, off offset:16"
                   :: "v"(la), "v"(src) : "memory");
    } else {
      if (bm + arow < M) {
        bf16_t* dst = (bf16_t*)sAu[buf] + arow * (LDAU * 2) + acb;
        const AT* src = A + (size_t)(bm + arow) * K + k0 + acb;
        #pragma unroll
        for (int j = 0; j < 16; ++j) dst[j] = (bf16_t)((float)src[j]);
      }
    }
    // W tile: f32 -> bf16, K-pair packed per column
    {
      const float* src = W + (size_t)(k0 + brow) * N + bn + bcb;
      bf16_t* sBh = (bf16_t*)sBu[buf];
      #pragma unroll
      for (int j = 0; j < 8; ++j)
        sBh[(bcb + j) * (LDBU * 2) + brow] = (bf16_t)src[j];
      if (k0 + BK < K) __builtin_prefetch(src + (size_t)BK * N, 0, 0);
    }
  };

  v8f acc[4] = { v8f{}, v8f{}, v8f{}, v8f{} };

  // compute one K-step from LDS buffer `buf`
  auto compute = [&](int buf) {
    const unsigned int* sAc = sAu[buf];
    const unsigned int* sBc = sBu[buf];
    // A fragment: two b128 loads (ISA 16-bit A 16x32 layout)
    V16 a;
    const unsigned int* rp = &sAc[(mi * 16 + l16) * LDAU + lh * 4];
    a.q[0] = *(const u32x4*)(rp);
    a.q[1] = *(const u32x4*)(rp + 8);
    // all 4 B fragments (ds_load_b128 pairs), then 4 WMMA
    V16 b[4];
    #pragma unroll
    for (int t = 0; t < 4; ++t) {
      const unsigned int* cp = &sBc[(t * 16 + l16) * LDBU + lh * 4];
      b[t].q[0] = *(const u32x4*)(cp);
      b[t].q[1] = *(const u32x4*)(cp + 8);
    }
    #pragma unroll
    for (int t = 0; t < 4; ++t)
      acc[t] = __builtin_amdgcn_wmma_f32_16x16x32_bf16(
          false, a.v, false, b[t].v, (short)0, acc[t], false, false);
  };

  const int steps = K / BK;

  stage(0, 0);
  if constexpr (kAsyncA) asm volatile("s_wait_asynccnt 0" ::: "memory");
  __syncthreads();

  // branch-free pipelined loop; last iteration peeled
  for (int i = 0; i < steps - 1; ++i) {
    const int cur = i & 1;
    stage(1 - cur, (i + 1) * BK);          // overlap with compute below
    compute(cur);
    if constexpr (kAsyncA) asm volatile("s_wait_asynccnt 0" ::: "memory");
    __syncthreads();
  }
  compute((steps - 1) & 1);

  // ---- epilogue (C layout: VGPR r -> M = 8*lh + r, N = l16) ----
  #pragma unroll
  for (int t = 0; t < 4; ++t) {
    const int col = bn + t * 16 + l16;
    const float bv = bias ? bias[col] : 0.0f;
    #pragma unroll
    for (int r = 0; r < 8; ++r) {
      const int row = bm + mi * 16 + lh * 8 + r;
      if (kAsyncA || row < M) {        // bf16-A path: M%128==0, guard folds away
        float x = acc[t][r] + bv;
        if (GELU) {
          float x3 = x * x * x;
          x = 0.5f * x * (1.0f + tanhf(0.7978845608028654f * (x + 0.044715f * x3)));
        }
        if (residual) x += residual[(size_t)row * N + col];
        C[(size_t)row * N + col] = (OutT)x;
      }
    }
  }
}

// ---------------------------------------------------------------------------
// Varlen cross-attention. One workgroup per batch element.
// ---------------------------------------------------------------------------
__global__ __launch_bounds__(256)
void cross_attn_kernel(const bf16_t* __restrict__ Q, const bf16_t* __restrict__ Kt,
                       const bf16_t* __restrict__ Vt, const int* __restrict__ cap,
                       bf16_t* __restrict__ O)
{
  __shared__ float sP[BQ * 80];
  const int b   = blockIdx.x;
  const int tid = threadIdx.x;
  const int nvalid = cap[b];
  const bf16_t* Kb = Kt + (size_t)b * LSEQ * DIM;
  const bf16_t* Vb = Vt + (size_t)b * LSEQ * DIM;

  // logits (128-bit bf16 loads)
  for (int idx = tid; idx < BQ * LSEQ; idx += 256) {
    int q = idx / LSEQ, l = idx - q * LSEQ;
    const bf16x8* qp = (const bf16x8*)(Q  + (size_t)q * DIM);
    const bf16x8* kp = (const bf16x8*)(Kb + (size_t)l * DIM);
    float acc = 0.0f;
    for (int d8 = 0; d8 < DIM / 8; ++d8) {
      bf16x8 qv = qp[d8], kv = kp[d8];
      #pragma unroll
      for (int e = 0; e < 8; ++e) acc += (float)qv[e] * (float)kv[e];
    }
    sP[q * 80 + l] = (l < nvalid) ? acc * 0.125f : -1e9f;
  }
  __syncthreads();

  // softmax rows (one thread per query)
  if (tid < BQ) {
    float* row = &sP[tid * 80];
    float m = row[0];
    for (int l = 1; l < LSEQ; ++l) m = fmaxf(m, row[l]);
    float s = 0.0f;
    for (int l = 0; l < LSEQ; ++l) { float e = __expf(row[l] - m); row[l] = e; s += e; }
    float inv = 1.0f / s;
    for (int l = 0; l < LSEQ; ++l) row[l] *= inv;
  }
  __syncthreads();

  // O = P @ V : wave per query, lanes own bf16 pairs across the 512-dim
  const int wave = tid >> 5, lane = tid & 31;
  for (int q = wave; q < BQ; q += 8) {
    float a0[8], a1[8];
    #pragma unroll
    for (int j = 0; j < 8; ++j) { a0[j] = 0.0f; a1[j] = 0.0f; }
    const float* p = &sP[q * 80];
    for (int l = 0; l < LSEQ; ++l) {
      float pl = p[l];
      const bf16x2* vp = (const bf16x2*)(Vb + (size_t)l * DIM);
      #pragma unroll
      for (int j = 0; j < 8; ++j) {
        bf16x2 vv = vp[lane + j * 32];
        a0[j] += pl * (float)vv[0];
        a1[j] += pl * (float)vv[1];
      }
    }
    bf16x2* op = (bf16x2*)(O + ((size_t)b * BQ + q) * DIM);
    #pragma unroll
    for (int j = 0; j < 8; ++j) {
      bf16x2 ov; ov[0] = (bf16_t)a0[j]; ov[1] = (bf16_t)a1[j];
      op[lane + j * 32] = ov;
    }
  }
}

// ---------------------------------------------------------------------------
// Self-attention per (batch, head). QKV [B,64,1536] bf16 (q|k|v each 512).
// ---------------------------------------------------------------------------
__global__ __launch_bounds__(256)
void self_attn_kernel(const bf16_t* __restrict__ QKV, bf16_t* __restrict__ O)
{
  __shared__ float sQ[BQ * HEADD], sK[BQ * HEADD], sP[BQ * BQ];
  const int b = blockIdx.x, h = blockIdx.y;
  const int tid = threadIdx.x;
  const bf16_t* base = QKV + (size_t)b * BQ * (3 * DIM);
  const int hc = h * HEADD;

  for (int idx = tid; idx < BQ * HEADD; idx += 256) {
    int r = idx >> 6, c = idx & 63;
    sQ[idx] = (float)base[(size_t)r * (3 * DIM) + hc + c];
    sK[idx] = (float)base[(size_t)r * (3 * DIM) + DIM + hc + c];
  }
  __syncthreads();

  for (int idx = tid; idx < BQ * BQ; idx += 256) {
    int q = idx >> 6, k = idx & 63;
    const float* qp = &sQ[q * HEADD];
    const float* kp = &sK[k * HEADD];
    float acc = 0.0f;
    #pragma unroll
    for (int d = 0; d < HEADD; ++d) acc += qp[d] * kp[d];
    sP[idx] = acc * 0.125f;
  }
  __syncthreads();

  if (tid < BQ) {
    float* row = &sP[tid * BQ];
    float m = row[0];
    for (int k = 1; k < BQ; ++k) m = fmaxf(m, row[k]);
    float s = 0.0f;
    for (int k = 0; k < BQ; ++k) { float e = __expf(row[k] - m); row[k] = e; s += e; }
    float inv = 1.0f / s;
    for (int k = 0; k < BQ; ++k) row[k] *= inv;
  }
  __syncthreads();

  const int wave = tid >> 5, lane = tid & 31;
  for (int q = wave; q < BQ; q += 8) {
    float a0 = 0.0f, a1 = 0.0f;
    const float* p = &sP[q * BQ];
    for (int k = 0; k < BQ; ++k) {
      float pk = p[k];
      const bf16x2* vp = (const bf16x2*)(base + (size_t)k * (3 * DIM) + 2 * DIM + hc);
      bf16x2 vv = vp[lane];
      a0 += pk * (float)vv[0];
      a1 += pk * (float)vv[1];
    }
    bf16x2* op = (bf16x2*)(O + ((size_t)b * BQ + q) * DIM + hc);
    bf16x2 ov; ov[0] = (bf16_t)a0; ov[1] = (bf16_t)a1;
    op[lane] = ov;
  }
}

// ---------------------------------------------------------------------------
// LayerNorm over rows of 512. One wave32 per row (16 f32 per lane).
// ---------------------------------------------------------------------------
template<typename OutT>
__global__ __launch_bounds__(256)
void layernorm_kernel(const float* __restrict__ X, const float* __restrict__ gamma,
                      const float* __restrict__ beta, OutT* __restrict__ Y, int rows)
{
  const int wave = threadIdx.x >> 5, lane = threadIdx.x & 31;
  const int row = blockIdx.x * 8 + wave;
  if (row >= rows) return;
  const float* x = X + (size_t)row * DIM;
  float vals[16], s = 0.0f, s2 = 0.0f;
  #pragma unroll
  for (int j = 0; j < 16; ++j) {
    float t = x[lane + j * 32];
    vals[j] = t; s += t; s2 += t * t;
  }
  s = wave_sum(s); s2 = wave_sum(s2);
  const float mean = s * (1.0f / DIM);
  const float var  = s2 * (1.0f / DIM) - mean * mean;
  const float inv  = rsqrtf(var + 1e-5f);
  OutT* y = Y + (size_t)row * DIM;
  #pragma unroll
  for (int j = 0; j < 16; ++j) {
    int d = lane + j * 32;
    y[d] = (OutT)((vals[j] - mean) * inv * gamma[d] + beta[d]);
  }
}

// x[b,q,d] += pos[q,d]   (64*512 = 32768 is a power of two)
__global__ __launch_bounds__(256)
void add_pos_kernel(float* __restrict__ X, const float* __restrict__ pos, size_t n)
{
  size_t i = (size_t)blockIdx.x * 256 + threadIdx.x;
  if (i < n) X[i] += pos[i & 32767u];
}

// ---------------------------------------------------------------------------
// Host-side orchestration
// ---------------------------------------------------------------------------
extern "C" void kernel_launch(void* const* d_in, const int* in_sizes, int n_in,
                              void* d_out, int out_size, void* d_ws, size_t ws_size,
                              hipStream_t stream)
{
  (void)in_sizes; (void)n_in; (void)out_size; (void)ws_size;

  const float* te      = (const float*)d_in[0];
  const int*   cap     = (const int*)  d_in[1];
  const float* in_w    = (const float*)d_in[2];
  const float* in_b    = (const float*)d_in[3];
  const float* mq      = (const float*)d_in[4];
  const float* wq      = (const float*)d_in[5];
  const float* wk      = (const float*)d_in[6];
  const float* wv      = (const float*)d_in[7];
  const float* cproj_w = (const float*)d_in[8];
  const float* cproj_b = (const float*)d_in[9];
  const float* pos     = (const float*)d_in[10];
  const float* ln1_s   = (const float*)d_in[11];
  const float* ln1_b   = (const float*)d_in[12];
  const float* qkv_w   = (const float*)d_in[13];
  const float* qkv_b   = (const float*)d_in[14];
  const float* aproj_w = (const float*)d_in[15];
  const float* aproj_b = (const float*)d_in[16];
  const float* ln2_s   = (const float*)d_in[17];
  const float* ln2_b   = (const float*)d_in[18];
  const float* fc1_w   = (const float*)d_in[19];
  const float* fc1_b   = (const float*)d_in[20];
  const float* fc2_w   = (const float*)d_in[21];
  const float* fc2_b   = (const float*)d_in[22];
  const float* lnf_s   = (const float*)d_in[23];
  const float* lnf_b   = (const float*)d_in[24];

  const int ML = NB * LSEQ;   // 78848 = 128 * 616
  const int MX = NB * BQ;     // 65536 = 128 * 512

  // ---- workspace layout (bytes). Peak ~450 MB with phase reuse. ----
  char* ws = (char*)d_ws;
  float*  x   = (float*)ws;                                    // [MX,512] f32
  char*   R   = ws + (size_t)MX * DIM * sizeof(float);         // reused region
  // phase A
  bf16_t* feat = (bf16_t*)(R);                                 // [ML,512]
  bf16_t* kbuf = (bf16_t*)(R + (size_t)ML * DIM * 2);
  bf16_t* vbuf = (bf16_t*)(R + (size_t)ML * DIM * 4);
  bf16_t* qbuf = (bf16_t*)(R + (size_t)ML * DIM * 6);          // [64,512]
  bf16_t* xatt = (bf16_t*)(R + (size_t)ML * DIM * 6 + BQ * DIM * 2);
  // phase B (overwrites phase A region)
  bf16_t* ybuf = (bf16_t*)(R);                                 // [MX,512]
  bf16_t* qkv  = (bf16_t*)(R + (size_t)MX * DIM * 2);          // [MX,1536]
  bf16_t* oatt = (bf16_t*)(R + (size_t)MX * DIM * 2 + (size_t)MX * 3 * DIM * 2);
  bf16_t* h1   = qkv;                                          // [MX,1024] reuse

  dim3 blk(256);
  #define GGRID(M_, N_) dim3(((M_) + 127) / 128, (N_) / 64)

  // 1) feat = text_embed @ in_w + in_b           (78848 x 768 x 512)
  gemm_wmma<float, bf16_t, false><<<GGRID(ML, DIM), blk, 0, stream>>>(
      te, in_w, in_b, nullptr, feat, ML, DIM, TXTD);
  // 2) k = feat @ wk ; v = feat @ wv             (78848 x 512 x 512)
  gemm_wmma<bf16_t, bf16_t, false><<<GGRID(ML, DIM), blk, 0, stream>>>(
      feat, wk, nullptr, nullptr, kbuf, ML, DIM, DIM);
  gemm_wmma<bf16_t, bf16_t, false><<<GGRID(ML, DIM), blk, 0, stream>>>(
      feat, wv, nullptr, nullptr, vbuf, ML, DIM, DIM);
  // 3) q = mq @ wq                               (64 x 512 x 512, row-guarded)
  gemm_wmma<float, bf16_t, false><<<GGRID(BQ, DIM), blk, 0, stream>>>(
      mq, wq, nullptr, nullptr, qbuf, BQ, DIM, DIM);
  // 4) varlen cross-attention
  cross_attn_kernel<<<dim3(NB), blk, 0, stream>>>(qbuf, kbuf, vbuf, cap, xatt);
  // 5) x = xatt @ cproj_w + cproj_b              (65536 x 512 x 512)
  gemm_wmma<bf16_t, float, false><<<GGRID(MX, DIM), blk, 0, stream>>>(
      xatt, cproj_w, cproj_b, nullptr, x, MX, DIM, DIM);
  // 6) x += pos
  {
    size_t n = (size_t)MX * DIM;
    add_pos_kernel<<<dim3((unsigned)((n + 255) / 256)), blk, 0, stream>>>(x, pos, n);
  }

  // 7) transformer blocks
  for (int i = 0; i < 3; ++i) {
    layernorm_kernel<bf16_t><<<dim3(MX / 8), blk, 0, stream>>>(
        x, ln1_s + i * DIM, ln1_b + i * DIM, ybuf, MX);
    gemm_wmma<bf16_t, bf16_t, false><<<GGRID(MX, 3 * DIM), blk, 0, stream>>>(
        ybuf, qkv_w + (size_t)i * DIM * 3 * DIM, qkv_b + i * 3 * DIM, nullptr,
        qkv, MX, 3 * DIM, DIM);
    self_attn_kernel<<<dim3(NB, NH), blk, 0, stream>>>(qkv, oatt);
    gemm_wmma<bf16_t, float, false><<<GGRID(MX, DIM), blk, 0, stream>>>(
        oatt, aproj_w + (size_t)i * DIM * DIM, aproj_b + i * DIM, x,   // +residual
        x, MX, DIM, DIM);
    layernorm_kernel<bf16_t><<<dim3(MX / 8), blk, 0, stream>>>(
        x, ln2_s + i * DIM, ln2_b + i * DIM, ybuf, MX);
    gemm_wmma<bf16_t, bf16_t, true><<<GGRID(MX, MLPD), blk, 0, stream>>>(
        ybuf, fc1_w + (size_t)i * DIM * MLPD, fc1_b + i * MLPD, nullptr,
        h1, MX, MLPD, DIM);
    gemm_wmma<bf16_t, float, false><<<GGRID(MX, DIM), blk, 0, stream>>>(
        h1, fc2_w + (size_t)i * MLPD * DIM, fc2_b + i * DIM, x,        // +residual
        x, MX, DIM, MLPD);
  }

  // 8) final LayerNorm -> d_out (f32)
  layernorm_kernel<float><<<dim3(MX / 8), blk, 0, stream>>>(
      x, lnf_s, lnf_b, (float*)d_out, MX);
}
